// TritonDynamicAttention_30081950941905
// MI455X (gfx1250) — compile-verified, matrix operand order checked by reading, not run on hardware
//
#include <hip/hip_runtime.h>

#define BATCH   8
#define HEADS   16
#define SEQ     1024
#define HDIM    64
#define QT      16      // q rows per workgroup (one wave32)
#define KSTRIDE 72      // halves: K LDS row stride (144B, 16B-aligned, conflict-free)
#define VSTRIDE 40      // halves: V^T LDS row stride (80B, 16B-aligned)
#define PSTRIDE 40      // halves: P LDS row stride

typedef __attribute__((ext_vector_type(16))) _Float16 v16h;
typedef __attribute__((ext_vector_type(8)))  _Float16 v8h;
typedef __attribute__((ext_vector_type(4)))  _Float16 v4h;
typedef __attribute__((ext_vector_type(8)))  float    v8f;
typedef __attribute__((ext_vector_type(4)))  float    v4f;

static __device__ __forceinline__ v16h cat8(v8h lo, v8h hi) {
  return __builtin_shufflevector(lo, hi, 0,1,2,3,4,5,6,7,8,9,10,11,12,13,14,15);
}

static __device__ __forceinline__ v8f wmma_f16(v16h a, v16h b, v8f c) {
  // D = A(16x32 f16) * B(32x16 f16) + C(16x16 f32)
  return __builtin_amdgcn_wmma_f32_16x16x32_f16(false, a, false, b, (short)0, c,
                                                false, false);
}

__global__ __launch_bounds__(32)
void blocksparse_attn_fwd(const float* __restrict__ Q,
                          const float* __restrict__ K,
                          const float* __restrict__ V,
                          const float* __restrict__ M,
                          float* __restrict__ O) {
  __shared__ _Float16 Kld[32 * KSTRIDE];    // K tile, row-major f16 [32][64]
  __shared__ _Float16 Vt [HDIM * VSTRIDE];  // V tile transposed f16 [64][32]
  __shared__ _Float16 Pld[QT * PSTRIDE];    // probs f16 [16][32]

  const int lane = threadIdx.x;   // 0..31
  const int g    = lane >> 4;     // half-wave
  const int m    = lane & 15;

  int blk = blockIdx.x;
  const int qt = blk & 63;  blk >>= 6;      // 64 q-tiles of 16 rows
  const int h  = blk & 15;  blk >>= 4;
  const int b  = blk;
  const int q0 = qt * QT;

  const size_t bh = (size_t)b * HEADS + h;
  const float* Qp = Q + (bh * SEQ + q0) * HDIM;
  const float* Kp = K + bh * SEQ * HDIM;
  const float* Vp = V + bh * SEQ * HDIM;
  const float* Mp = M + ((size_t)h * SEQ + q0) * SEQ;
  float*       Op = O + (bh * SEQ + q0) * HDIM;

  // ---- Q tile -> A-operand registers (16x64 f16 as two K=32 chunks) ----
  // A layout: lanes 0-15 row m holds K {0..7, 16..23}; lanes 16-31 hold {8..15, 24..31}
  v16h aq[2];
  {
    const float* qrow = Qp + (size_t)m * HDIM;
    #pragma unroll
    for (int kc = 0; kc < 2; ++kc) {
      const int b0 = kc * 32 + g * 8;
      v16h a;
      #pragma unroll
      for (int i = 0; i < 8; ++i) a[i]     = (_Float16)qrow[b0 + i];
      #pragma unroll
      for (int i = 0; i < 8; ++i) a[8 + i] = (_Float16)qrow[b0 + 16 + i];
      aq[kc] = a;
    }
  }

  // all-ones B operand: D = P * 1 gives softmax row-sums, replicated per column,
  // in the same C-layout as the o-accumulators (replaces shuffle-based row sums)
  v16h bones;
  #pragma unroll
  for (int i = 0; i < 16; ++i) bones[i] = (_Float16)1.0f;

  float mrun[8], alpha[8];
  v8f o0 = {}, o1 = {}, o2 = {}, o3 = {};
  v8f oL = {};                                   // running softmax denominator l
  #pragma unroll
  for (int r = 0; r < 8; ++r) mrun[r] = -__builtin_inff();

  const int jdiag = q0 >> 5;                     // causal block-column limit
  for (int kb = 0; kb <= jdiag; ++kb) {
    // Block-mask test: one uniform scalar read; SGPR branch keeps EXEC all-ones.
    const int pb = __builtin_amdgcn_readfirstlane(__float_as_int(Mp[kb * 32]));
    if (pb == 0) continue;
    const int k0 = kb * 32;

    __syncthreads();  // previous iteration's LDS reads complete before overwrite
    // ---- stage K (row-major f16) into LDS ----
    #pragma unroll 4
    for (int it = 0; it < 16; ++it) {
      const int gi  = it * 32 + lane;   // 512 float4 groups
      const int row = gi >> 4;
      const int c4  = gi & 15;
      v4f kq = *(const v4f*)(Kp + (size_t)(k0 + row) * HDIM + c4 * 4);
      v4h kh;
      #pragma unroll
      for (int i = 0; i < 4; ++i) kh[i] = (_Float16)kq[i];
      *(v4h*)(&Kld[row * KSTRIDE + c4 * 4]) = kh;
    }
    // ---- stage V transposed via 4x4 register transpose (b64 LDS stores) ----
    {
      const int k4 = (lane & 7) * 4;          // k rows handled by this lane
      const int d4 = (lane >> 3) * 4;         // base dim chunk
      #pragma unroll
      for (int t = 0; t < 4; ++t) {
        const int d0 = d4 + 16 * t;
        const float* vbase = Vp + (size_t)(k0 + k4) * HDIM + d0;
        v4f r0 = *(const v4f*)(vbase);
        v4f r1 = *(const v4f*)(vbase + HDIM);
        v4f r2 = *(const v4f*)(vbase + 2 * HDIM);
        v4f r3 = *(const v4f*)(vbase + 3 * HDIM);
        #pragma unroll
        for (int i = 0; i < 4; ++i) {
          v4h tv;
          tv[0] = (_Float16)r0[i]; tv[1] = (_Float16)r1[i];
          tv[2] = (_Float16)r2[i]; tv[3] = (_Float16)r3[i];
          *(v4h*)(&Vt[(d0 + i) * VSTRIDE + k4]) = tv;
        }
      }
    }
    __syncthreads();

    // ---- S = Q K^T : 16x32 scores, two N-tiles, two K=32 chunks -> 4 WMMAs ----
    v8f s0 = {}, s1 = {};
    #pragma unroll
    for (int kc = 0; kc < 2; ++kc) {
      const _Float16* kr0 = &Kld[( 0 + m) * KSTRIDE + kc * 32 + g * 8];
      const _Float16* kr1 = &Kld[(16 + m) * KSTRIDE + kc * 32 + g * 8];
      v16h bk0 = cat8(*(const v8h*)kr0, *(const v8h*)(kr0 + 16));
      v16h bk1 = cat8(*(const v8h*)kr1, *(const v8h*)(kr1 + 16));
      s0 = wmma_f16(aq[kc], bk0, s0);
      s1 = wmma_f16(aq[kc], bk1, s1);
    }

    // ---- element-wise causal mask on the diagonal block ----
    if (kb == jdiag) {
      #pragma unroll
      for (int r = 0; r < 8; ++r) {
        const int row = q0 + r + 8 * g;
        s0[r] = (k0 + m      > row) ? -__builtin_inff() : s0[r];
        s1[r] = (k0 + 16 + m > row) ? -__builtin_inff() : s1[r];
      }
    }

    // ---- online softmax: row max via 16-lane butterflies, exp in f32 ----
    #pragma unroll
    for (int r = 0; r < 8; ++r) {
      float t = fmaxf(s0[r], s1[r]);
      t = fmaxf(t, __shfl_xor(t, 1, 32));
      t = fmaxf(t, __shfl_xor(t, 2, 32));
      t = fmaxf(t, __shfl_xor(t, 4, 32));
      t = fmaxf(t, __shfl_xor(t, 8, 32));
      const float mn = fmaxf(mrun[r], t);
      alpha[r] = __expf(mrun[r] - mn);
      mrun[r]  = mn;
      s0[r] = __expf(s0[r] - mn);
      s1[r] = __expf(s1[r] - mn);
    }
    #pragma unroll
    for (int r = 0; r < 8; ++r) {
      o0[r] *= alpha[r]; o1[r] *= alpha[r]; o2[r] *= alpha[r]; o3[r] *= alpha[r];
      oL[r] *= alpha[r];
    }

    // ---- P (C-layout) -> LDS row-major f16 -> A-layout reload ----
    #pragma unroll
    for (int r = 0; r < 8; ++r) {
      Pld[(r + 8 * g) * PSTRIDE + m]      = (_Float16)s0[r];
      Pld[(r + 8 * g) * PSTRIDE + 16 + m] = (_Float16)s1[r];
    }
    __syncthreads();

    const _Float16* pr = &Pld[m * PSTRIDE + g * 8];
    const v16h ap = cat8(*(const v8h*)pr, *(const v8h*)(pr + 16));

    // ---- O += P V (4 WMMAs) and l += rowsum(P) (1 WMMA against ones) ----
    {
      const _Float16* vr0 = &Vt[( 0 + m) * VSTRIDE + g * 8];
      const _Float16* vr1 = &Vt[(16 + m) * VSTRIDE + g * 8];
      const _Float16* vr2 = &Vt[(32 + m) * VSTRIDE + g * 8];
      const _Float16* vr3 = &Vt[(48 + m) * VSTRIDE + g * 8];
      o0 = wmma_f16(ap, cat8(*(const v8h*)vr0, *(const v8h*)(vr0 + 16)), o0);
      o1 = wmma_f16(ap, cat8(*(const v8h*)vr1, *(const v8h*)(vr1 + 16)), o1);
      o2 = wmma_f16(ap, cat8(*(const v8h*)vr2, *(const v8h*)(vr2 + 16)), o2);
      o3 = wmma_f16(ap, cat8(*(const v8h*)vr3, *(const v8h*)(vr3 + 16)), o3);
      oL = wmma_f16(ap, bones, oL);
    }
  }

  // ---- epilogue: O / l (diagonal block guarantees l >= 1) ----
  #pragma unroll
  for (int r = 0; r < 8; ++r) {
    const float inv = 1.0f / oL[r];
    float* orow = Op + (size_t)(r + 8 * g) * HDIM;
    orow[ 0 + m] = o0[r] * inv;
    orow[16 + m] = o1[r] * inv;
    orow[32 + m] = o2[r] * inv;
    orow[48 + m] = o3[r] * inv;
  }
}

extern "C" void kernel_launch(void* const* d_in, const int* in_sizes, int n_in,
                              void* d_out, int out_size, void* d_ws, size_t ws_size,
                              hipStream_t stream) {
  (void)in_sizes; (void)n_in; (void)out_size; (void)d_ws; (void)ws_size;
  const float* q = (const float*)d_in[0];
  const float* k = (const float*)d_in[1];
  const float* v = (const float*)d_in[2];
  const float* m = (const float*)d_in[3];
  float* out = (float*)d_out;
  const int grid = BATCH * HEADS * (SEQ / QT);   // 8192 workgroups, 1 wave each
  blocksparse_attn_fwd<<<dim3(grid), dim3(32), 0, stream>>>(q, k, v, m, out);
}
